// SpectrumEncoderRefine_35038343201093
// MI455X (gfx1250) — matrix operand order repaired and checked by reading, not compiled, not run on hardware
//
#include <hip/hip_runtime.h>
#include <hip/hip_bf16.h>
#include <math.h>
#include <stdint.h>

#define B_ 8
#define P_ 1023
#define N_ 1024
#define D_ 128
#define H_ 32
#define ALPHA_ 0.2f
#define NEGV_ -9e15f

typedef __attribute__((ext_vector_type(2)))  float          v2f;
typedef __attribute__((ext_vector_type(8)))  float          v8f;
typedef __attribute__((ext_vector_type(16))) unsigned short v16us;
typedef __attribute__((ext_vector_type(16))) __bf16         v16bf;

__device__ __forceinline__ unsigned short bf16_bits(float x) {
  unsigned int u = __builtin_bit_cast(unsigned int, x);
  u += 0x7FFFu + ((u >> 16) & 1u);   // round-to-nearest-even
  return (unsigned short)(u >> 16);
}

// ---------------------------------------------------------------------------
// Kernel A: x = concat(latent, spectra@peak_w + peak_b); mz_full; mask output
// ---------------------------------------------------------------------------
__global__ void build_x_kernel(const float* __restrict__ spectra,
                               const float* __restrict__ mz,
                               const float* __restrict__ latent,
                               const float* __restrict__ peak_w,
                               const float* __restrict__ peak_b,
                               float* __restrict__ x,
                               float* __restrict__ mzf,
                               float* __restrict__ mask_out) {
  int bx = blockIdx.x;
  int b = bx / N_;
  int i = bx - b * N_;
  int d = threadIdx.x;
  size_t row = (size_t)b * N_ + i;
  if (i == 0) {
    x[row * D_ + d] = latent[d];
    if (d == 0) { mzf[row] = 0.f; mask_out[row] = 0.f; }
  } else {
    int p = i - 1;
    float s0 = spectra[((size_t)b * P_ + p) * 2 + 0];
    float s1 = spectra[((size_t)b * P_ + p) * 2 + 1];
    x[row * D_ + d] = s0 * peak_w[d] + s1 * peak_w[D_ + d] + peak_b[d];
    if (d == 0) {
      mzf[row] = mz[(size_t)b * P_ + p];
      mask_out[row] = ((s0 + s1) == 0.f) ? 1.f : 0.f;
    }
  }
}

// ---------------------------------------------------------------------------
// Kernel B: Wh = x @ W via V_WMMA_F32_16X16X4_F32 (exact f32 accumulate).
// Writes Wh (f32, row-major) and Whbt (bf16, B-fragment-friendly layout:
//   Whbt[(b*32 + row/32)*4096 + col*32 + (row%32)] )
// ---------------------------------------------------------------------------
__global__ void __launch_bounds__(128) wh_kernel(const float* __restrict__ x,
                                                 const float* __restrict__ W,
                                                 float* __restrict__ Wh,
                                                 unsigned short* __restrict__ Whbt) {
  int bx   = blockIdx.x;        // 0..127
  int b    = bx >> 4;
  int t64  = bx & 15;
  int wave = threadIdx.x >> 5;
  int lane = threadIdx.x & 31;
  int half = lane >> 4;
  int ln   = lane & 15;
  int rowbase = t64 * 64 + wave * 16;
  size_t bN = (size_t)b * N_;

  v8f acc[8] = {};
  const float* xrow = x + (bN + rowbase + ln) * D_;

  for (int kb = 0; kb < D_; kb += 4) {
    v2f av;                                    // A: lane=row, elem v -> K=v+2*half
    av.x = xrow[kb + 0 + 2 * half];
    av.y = xrow[kb + 1 + 2 * half];
#pragma unroll
    for (int ct = 0; ct < 8; ++ct) {
      v2f bv;                                  // B: lane=col, elem v -> K=v+2*half
      bv.x = W[(size_t)(kb + 0 + 2 * half) * D_ + ct * 16 + ln];
      bv.y = W[(size_t)(kb + 1 + 2 * half) * D_ + ct * 16 + ln];
      acc[ct] = __builtin_amdgcn_wmma_f32_16x16x4_f32(
          false, av, false, bv, (short)0, acc[ct], false, false);
    }
  }
#pragma unroll
  for (int ct = 0; ct < 8; ++ct) {
#pragma unroll
    for (int r = 0; r < 8; ++r) {
      int row = rowbase + r + 8 * half;        // C/D layout: VGPR r -> M=r, r+8
      int col = ct * 16 + ln;
      float v = acc[ct][r];
      Wh[(bN + row) * D_ + col] = v;
      Whbt[((size_t)b * 32 + (row >> 5)) * (D_ * 32) + col * 32 + (row & 31)] =
          bf16_bits(v);
    }
  }
}

// ---------------------------------------------------------------------------
// Kernel C: e1 = Wh @ a[:D], e2 = Wh @ a[D:]
// ---------------------------------------------------------------------------
__global__ void e12_kernel(const float* __restrict__ Wh,
                           const float* __restrict__ a,
                           float* __restrict__ e1, float* __restrict__ e2) {
  __shared__ float r1[128];
  __shared__ float r2[128];
  int bx = blockIdx.x;          // = b*N + i
  int d = threadIdx.x;
  float v = Wh[(size_t)bx * D_ + d];
  r1[d] = v * a[d];
  r2[d] = v * a[D_ + d];
  __syncthreads();
  for (int s = 64; s > 0; s >>= 1) {
    if (d < s) { r1[d] += r1[d + s]; r2[d] += r2[d + s]; }
    __syncthreads();
  }
  if (d == 0) { e1[bx] = r1[0]; e2[bx] = r2[0]; }
}

// ---------------------------------------------------------------------------
// Kernel D: fused GAT attention (flash style).
// grid = B*64 WGs (16 rows each), block = 128 (4 waves).
// Per iteration each wave:
//   - issues GLOBAL_LOAD_ASYNC_TO_LDS_B128 for its 32x128 bf16 Wh tile (8KB),
//     overlapped with the VALU-heavy pairwise MLP logits,
//   - online softmax, then s_wait_asynccnt 0 and P@Wh via WMMA bf16
//     with 2x ds_load_b128 per fragment.
// Deterministic 4-way wave combine through LDS (no float atomics).
// ---------------------------------------------------------------------------
__global__ void __launch_bounds__(128) gat_attn_kernel(
    const float* __restrict__ e1g, const float* __restrict__ e2g,
    const float* __restrict__ mzf, const int* __restrict__ adj,
    const unsigned short* __restrict__ Whbt,
    const float* __restrict__ mlp_w1, const float* __restrict__ mlp_b1,
    const float* __restrict__ mlp_w2, const float* __restrict__ mlp_b2,
    float* __restrict__ out) {
  // arena: [0,32K) staging (loop) / O-partials (epilogue); [32K,40K) logits;
  //        [40K,44K) bf16 probabilities
  __shared__ __align__(16) unsigned char arena[45056];
  unsigned short* s_whb = (unsigned short*)arena;                       // [4][4096]
  float (*s_s)[16][32] = (float (*)[16][32])(arena + 32768);            // [4][16][32]
  unsigned short (*s_p)[16][32] =
      (unsigned short (*)[16][32])(arena + 40960);                      // [4][16][32]
  float* s_O = (float*)arena;                                           // [4][2048]

  __shared__ float s_scale[4][16];
  __shared__ float s_m[4][16];
  __shared__ float s_l[4][16];
  __shared__ float s_gm[16];
  __shared__ float s_lc[16];
  __shared__ float s_e1[16];
  __shared__ float s_mzi[16];
  __shared__ float s_w1[H_], s_b1[H_], s_w2[H_];
  __shared__ float s_b2;

  int bx    = blockIdx.x;       // 0..511
  int b     = bx >> 6;
  int itile = bx & 63;
  int i0    = itile * 16;
  int tid   = threadIdx.x;
  int w     = tid >> 5;
  int lane  = tid & 31;
  int half  = lane >> 4;
  int ln    = lane & 15;
  size_t bN = (size_t)b * N_;

  if (tid < 16) { s_e1[tid] = e1g[bN + i0 + tid]; s_mzi[tid] = mzf[bN + i0 + tid]; }
  if (tid < H_) { s_w1[tid] = mlp_w1[tid]; s_b1[tid] = mlp_b1[tid]; s_w2[tid] = mlp_w2[tid]; }
  if (tid == 0) s_b2 = mlp_b2[0];
  __syncthreads();

  v8f acc[8] = {};
  float m_i = -__builtin_inff();
  float l_i = 0.f;

  // LDS byte offset of this wave's staging slice (low 32 bits of flat addr)
  unsigned lds_slice = (unsigned)(uintptr_t)(const void*)(s_whb) + (unsigned)(w * 8192);

  for (int it = 0; it < 8; ++it) {
    int jt = w + 4 * it;        // this wave's j-tile
    int j0 = jt * 32;
    int c  = lane;
    int jg = j0 + c;

    // ---- issue async copy of the 32x128 bf16 Wh tile into LDS ----
    {
      const unsigned char* gtile =
          (const unsigned char*)(Whbt + ((size_t)b * 32 + jt) * 4096);
#pragma unroll
      for (int t = 0; t < 16; ++t) {
        unsigned la = lds_slice + (unsigned)((t * 32 + lane) * 16);
        unsigned long long ga =
            (unsigned long long)(uintptr_t)(gtile + (t * 32 + lane) * 16);
        asm volatile("global_load_async_to_lds_b128 %0, %1, off"
                     :: "v"(la), "v"(ga) : "memory");
      }
    }

    float e2j = e2g[bN + jg];
    float mzj = mzf[bN + jg];
    if (it + 1 < 8)             // hint next adj tile toward the caches
      __builtin_prefetch(&adj[(bN + i0 + ln) * N_ + j0 + 128], 0, 0);

    // ---- phase 1: logits for 16x32 tile (overlaps the async copy) ----
    float dmz[16], t16[16];
    int   av[16];
#pragma unroll
    for (int i = 0; i < 16; ++i) {
      int ig = i0 + i;
      av[i]  = adj[(bN + ig) * N_ + jg];
      dmz[i] = (ig == 0 || jg == 0) ? 0.f : fabsf(s_mzi[i] - mzj);
      t16[i] = s_b2;
    }
#pragma unroll
    for (int k = 0; k < H_; ++k) {
      float w1k = s_w1[k], b1k = s_b1[k], w2k = s_w2[k];
#pragma unroll
      for (int i = 0; i < 16; ++i)
        t16[i] += fmaxf(dmz[i] * w1k + b1k, 0.f) * w2k;
    }
#pragma unroll
    for (int i = 0; i < 16; ++i) {
      float ev = s_e1[i] + e2j;
      ev = (ev >= 0.f) ? ev : ALPHA_ * ev;
      s_s[w][i][c] = (av[i] > 0) ? (ev + t16[i]) : NEGV_;
    }
    __syncthreads();

    // ---- phase 2: per-row online softmax (lanes 0..15 own rows) ----
    if (lane < 16) {
      int i = lane;
      float rmax = -__builtin_inff();
#pragma unroll
      for (int cc = 0; cc < 32; ++cc) rmax = fmaxf(rmax, s_s[w][i][cc]);
      float mnew  = fmaxf(m_i, rmax);
      float scale = __expf(m_i - mnew);
      float psum  = 0.f;
#pragma unroll
      for (int cc = 0; cc < 32; ++cc) {
        float p = __expf(s_s[w][i][cc] - mnew);
        psum += p;
        s_p[w][i][cc] = bf16_bits(p);
      }
      l_i = l_i * scale + psum;
      m_i = mnew;
      s_scale[w][i] = scale;
    }
    __syncthreads();

    // ---- phase 3: rescale accumulators by per-row factor ----
#pragma unroll
    for (int ct = 0; ct < 8; ++ct)
#pragma unroll
      for (int r = 0; r < 8; ++r)
        acc[ct][r] *= s_scale[w][r + 8 * half];

    // ---- phase 4: acc += P(16x32) @ Wh(32x128) via WMMA bf16 ----
    // A fragment: lane ln = row; elements are two contiguous 16B runs of [row][K]
    union Frag { v16us v; uint4 q[2]; };
    Frag fa;
    fa.q[0] = *(const uint4*)&s_p[w][ln][half * 16];
    fa.q[1] = *(const uint4*)&s_p[w][ln][half * 16 + 8];
    v16bf A = __builtin_bit_cast(v16bf, fa.v);

    // wait for the async-staged tile before reading it
    asm volatile("s_wait_asynccnt 0x0" ::: "memory");

    const unsigned short* tile = s_whb + w * 4096;   // [col][row] ushort
#pragma unroll
    for (int ct = 0; ct < 8; ++ct) {
      // B fragment: lane ln = col; K = half*16 + e -> 32 contiguous bytes
      Frag fb;
      const uint4* bp = (const uint4*)(tile + (ct * 16 + ln) * 32 + half * 16);
      fb.q[0] = bp[0];
      fb.q[1] = bp[1];
      v16bf Bf = __builtin_bit_cast(v16bf, fb.v);
      acc[ct] = __builtin_amdgcn_wmma_f32_16x16x32_bf16(
          false, A, false, Bf, (short)0, acc[ct], false, false);
    }
    __syncthreads();
  }

  // ---- deterministic 4-wave combine ----
  if (lane < 16) { s_m[w][lane] = m_i; s_l[w][lane] = l_i; }
  __syncthreads();
  if (w == 0 && lane < 16) {
    int i = lane;
    float gm = s_m[0][i];
    for (int ww = 1; ww < 4; ++ww) gm = fmaxf(gm, s_m[ww][i]);
    float lc = 0.f;
    for (int ww = 0; ww < 4; ++ww) lc += s_l[ww][i] * __expf(s_m[ww][i] - gm);
    s_gm[i] = gm;
    s_lc[i] = lc;
  }
  __syncthreads();
#pragma unroll
  for (int ct = 0; ct < 8; ++ct)
#pragma unroll
    for (int r = 0; r < 8; ++r) {
      int row = r + 8 * half;
      float f = __expf(s_m[w][row] - s_gm[row]);
      s_O[w * 2048 + row * 128 + ct * 16 + ln] = acc[ct][r] * f;
    }
  __syncthreads();
  for (int t = tid; t < 16 * 128; t += 128) {
    int row = t >> 7;
    float v = (s_O[t] + s_O[2048 + t] + s_O[4096 + t] + s_O[6144 + t]) / s_lc[row];
    out[(bN + i0 + row) * D_ + (t & 127)] = (v > 0.f) ? v : expm1f(v);
  }
}

// ---------------------------------------------------------------------------
extern "C" void kernel_launch(void* const* d_in, const int* in_sizes, int n_in,
                              void* d_out, int out_size, void* d_ws, size_t ws_size,
                              hipStream_t stream) {
  (void)in_sizes; (void)n_in; (void)out_size; (void)ws_size;
  const float* spectra = (const float*)d_in[0];
  const float* mz      = (const float*)d_in[1];
  const int*   adj     = (const int*)d_in[2];
  const float* latent  = (const float*)d_in[3];
  const float* peak_w  = (const float*)d_in[4];
  const float* peak_b  = (const float*)d_in[5];
  const float* W       = (const float*)d_in[6];
  const float* a       = (const float*)d_in[7];
  const float* mlp_w1  = (const float*)d_in[8];
  const float* mlp_b1  = (const float*)d_in[9];
  const float* mlp_w2  = (const float*)d_in[10];
  const float* mlp_b2  = (const float*)d_in[11];

  char* ws = (char*)d_ws;
  float* x            = (float*)ws;          ws += (size_t)B_ * N_ * D_ * 4;
  float* Wh           = (float*)ws;          ws += (size_t)B_ * N_ * D_ * 4;
  unsigned short* Whbt = (unsigned short*)ws; ws += (size_t)B_ * N_ * D_ * 2;
  float* mzf          = (float*)ws;          ws += (size_t)B_ * N_ * 4;
  float* e1           = (float*)ws;          ws += (size_t)B_ * N_ * 4;
  float* e2           = (float*)ws;          ws += (size_t)B_ * N_ * 4;

  float* out_x    = (float*)d_out;
  float* out_mask = out_x + (size_t)B_ * N_ * D_;

  build_x_kernel<<<B_ * N_, D_, 0, stream>>>(spectra, mz, latent, peak_w, peak_b,
                                             x, mzf, out_mask);
  wh_kernel<<<B_ * 16, 128, 0, stream>>>(x, W, Wh, Whbt);
  e12_kernel<<<B_ * N_, 128, 0, stream>>>(Wh, a, e1, e2);
  gat_attn_kernel<<<B_ * 64, 128, 0, stream>>>(e1, e2, mzf, adj, Whbt,
                                               mlp_w1, mlp_b1, mlp_w2, mlp_b2,
                                               out_x);
}
